// ConvLayer_44762149159176
// MI455X (gfx1250) — matrix operand order repaired and checked by reading, not compiled, not run on hardware
//
#include <hip/hip_runtime.h>
#include <hip/hip_bf16.h>
#include <math.h>

#define AF     64        // atom feature length
#define NB     41        // neighbor (edge) feature length
#define DIN    169       // 2*AF + NB
#define KP     192       // padded K (6 chunks of 32)
#define NCH    6         // K chunks of 32
#define MNBR   12        // neighbors per atom
#define NATOMS 100000
#define RSTR   200       // halfs per staged row (192 + 8 pad -> bank decorrelation)
#define WPB    8         // waves per block (256 threads)
#define EPSV   1e-5f

typedef __attribute__((ext_vector_type(16))) _Float16 v16h;
typedef __attribute__((ext_vector_type(8)))  _Float16 v8h;
typedef __attribute__((ext_vector_type(8)))  float    v8f;

// precise versions (used once per output element in k_out)
__device__ __forceinline__ float softplusf(float x) {
    return (x > 0.f) ? (x + log1pf(expf(-x))) : log1pf(expf(x));
}
// fast versions for the hot gating epilogue (hardware v_exp/v_log)
__device__ __forceinline__ float softplusf_fast(float x) {
    return (x > 20.f) ? x : __logf(1.f + __expf(x));
}
__device__ __forceinline__ float sigmoidf_fast(float x) {
    return 1.f / (1.f + __expf(-x));
}

// ---------------------------------------------------------------------------
// Stage the 12 real rows for atom n into LDS as f16:
// row r = [ self(64) | atom[idx[n][r]](64) | edge(41) | zeros..192 )
// Tail zeros (k in [169,192)) and pad rows 12..15 are zeroed once per wave
// before the atom loop (they never change).
// ---------------------------------------------------------------------------
__device__ __forceinline__ void zero_static(_Float16 (*rb)[RSTR], int lane) {
    #pragma unroll
    for (int r = 0; r < MNBR; ++r)
        for (int k = DIN + lane; k < KP; k += 32) rb[r][k] = (_Float16)0.f;
    #pragma unroll
    for (int r = MNBR; r < 16; ++r)
        for (int k = lane; k < KP; k += 32) rb[r][k] = (_Float16)0.f;
}

__device__ __forceinline__ void stage_rows(_Float16 (*rb)[RSTR],
        const float* __restrict__ atom, const float* __restrict__ nbrf,
        const int* __restrict__ idx, int n, int lane) {
    if (lane < 16) {
        // self features: identical for every row of this atom
        const float4 sv = *(const float4*)(atom + (size_t)n * AF + 4 * lane);
        #pragma unroll
        for (int r = 0; r < MNBR; ++r) {
            rb[r][4 * lane + 0] = (_Float16)sv.x;
            rb[r][4 * lane + 1] = (_Float16)sv.y;
            rb[r][4 * lane + 2] = (_Float16)sv.z;
            rb[r][4 * lane + 3] = (_Float16)sv.w;
        }
    } else {
        const int l = lane - 16;
        #pragma unroll
        for (int r = 0; r < MNBR; ++r) {
            const int mi = idx[n * MNBR + r];  // gather: hits L2 (table is 25.6MB)
            const float4 nv = *(const float4*)(atom + (size_t)mi * AF + 4 * l);
            rb[r][AF + 4 * l + 0] = (_Float16)nv.x;
            rb[r][AF + 4 * l + 1] = (_Float16)nv.y;
            rb[r][AF + 4 * l + 2] = (_Float16)nv.z;
            rb[r][AF + 4 * l + 3] = (_Float16)nv.w;
        }
    }
    // edge features (41 floats per row), lanes 0..10
    if (lane < 11) {
        #pragma unroll
        for (int r = 0; r < MNBR; ++r) {
            const float* ep = nbrf + ((size_t)n * MNBR + r) * NB;
            #pragma unroll
            for (int j = 0; j < 4; ++j) {
                const int k = 4 * lane + j;
                if (k < NB) rb[r][2 * AF + k] = (_Float16)ep[k];
            }
        }
    }
}

// ---------------------------------------------------------------------------
// 16x192 (rows x K) @ 192x128 (K x out) via 48 x v_wmma_f32_16x16x32_f16.
// A frags: lane L holds row L%16; lo lanes K {0..15}+32q,
// hi lanes +16.  B frags come pre-swizzled from global (one 32B load each).
// Accumulators start at 0 so the first WMMA of each tile uses inline SRC2=0
// (no splat movs); bias is folded into the epilogues.
// ---------------------------------------------------------------------------
__device__ __forceinline__ void gemm_tile(const _Float16 (*rb)[RSTR],
        const _Float16* __restrict__ bswz, int lane, v8f acc[8]) {
    const int r  = lane & 15;
    const int hi = lane >> 4;
    #pragma unroll
    for (int t = 0; t < 8; ++t) {
        #pragma unroll
        for (int e = 0; e < 8; ++e) acc[t][e] = 0.f;
    }
    v16h afrag[NCH];
    #pragma unroll
    for (int q = 0; q < NCH; ++q) {
        const v8h lo = *(const v8h*)&rb[r][32 * q + 16 * hi];
        const v8h hc = *(const v8h*)&rb[r][32 * q + 16 * hi + 8];
        afrag[q] = __builtin_shufflevector(lo, hc, 0,1,2,3,4,5,6,7,
                                                   8,9,10,11,12,13,14,15);
    }
    #pragma unroll
    for (int q = 0; q < NCH; ++q) {
        #pragma unroll
        for (int t = 0; t < 8; ++t) {
            const v16h bf = *(const v16h*)(bswz + (size_t)((q * 8 + t) * 32 + lane) * 16);
            acc[t] = __builtin_amdgcn_wmma_f32_16x16x32_f16(
                false, afrag[q], false, bf, (short)0, acc[t], false, false);
        }
    }
}

// ---------------------------------------------------------------------------
// Kernel 0: zero BN-stat accumulators; build swizzled f16 W^T fragments.
// Bswz[((q*8+t)*32+lane)*16 + kk] = W[16t+lane%16][32q + (lane>=16?16:0) + kk]
// ---------------------------------------------------------------------------
__global__ void k_init(const float* __restrict__ W, _Float16* __restrict__ bswz,
                       float* __restrict__ gstats1, float* __restrict__ gstats2) {
    const int gt = blockIdx.x * blockDim.x + threadIdx.x;
    const int nt = gridDim.x * blockDim.x;
    for (int i = gt; i < 256; i += nt) gstats1[i] = 0.f;
    for (int i = gt; i < 128; i += nt) gstats2[i] = 0.f;
    for (int i = gt; i < NCH * 8 * 32 * 16; i += nt) {
        const int kk = i & 15, ln = (i >> 4) & 31, t = (i >> 9) & 7, q = i >> 12;
        const int o = 16 * t + (ln & 15);
        const int k = 32 * q + ((ln >= 16) ? 16 : 0) + kk;
        bswz[i] = (k < DIN) ? (_Float16)W[o * DIN + k] : (_Float16)0.f;
    }
}

// ---------------------------------------------------------------------------
// Kernel 1: GEMM pass accumulating per-channel sum / sum-of-squares (with the
// bias added) over the 1.2M real rows (pad rows masked).  Register
// accumulation across the wave's atoms, LDS hop, then one global atomicAdd
// per channel per block.
// ---------------------------------------------------------------------------
__global__ __launch_bounds__(256) void k_stats1(
        const float* __restrict__ atom, const float* __restrict__ nbrf,
        const int* __restrict__ idx, const _Float16* __restrict__ bswz,
        const float* __restrict__ bias, float* __restrict__ gstats1) {
    __shared__ _Float16 rowbuf[WPB][16][RSTR];
    __shared__ float shacc[256];  // [0..127]=sum, [128..255]=sumsq
    const int tid = threadIdx.x, lane = tid & 31, w = tid >> 5;
    shacc[tid] = 0.f;
    __syncthreads();

    const int gw = blockIdx.x * WPB + w;
    const int nw = gridDim.x * WPB;
    const int hi = lane >> 4;

    float bc[8];                                  // loop-invariant bias per (t,lane)
    #pragma unroll
    for (int t = 0; t < 8; ++t) bc[t] = bias[16 * t + (lane & 15)];

    float rs[8], rq[8];
    #pragma unroll
    for (int t = 0; t < 8; ++t) { rs[t] = 0.f; rq[t] = 0.f; }

    zero_static(rowbuf[w], lane);

    for (int n = gw; n < NATOMS; n += nw) {
        stage_rows(rowbuf[w], atom, nbrf, idx, n, lane);
        asm volatile("s_wait_dscnt 0" ::: "memory");   // LDS stores visible to frag loads
        v8f acc[8];
        gemm_tile(rowbuf[w], bswz, lane, acc);
        #pragma unroll
        for (int t = 0; t < 8; ++t) {
            float s = 0.f, q2 = 0.f;
            #pragma unroll
            for (int e = 0; e < 8; ++e) {
                const int row = e + 8 * hi;
                if (row < MNBR) {
                    const float v = acc[t][e] + bc[t];
                    s += v; q2 += v * v;
                }
            }
            rs[t] += s; rq[t] += q2;
        }
    }
    #pragma unroll
    for (int t = 0; t < 8; ++t) {
        const float s  = rs[t] + __shfl_xor(rs[t], 16);
        const float q2 = rq[t] + __shfl_xor(rq[t], 16);
        if (lane < 16) {
            atomicAdd(&shacc[16 * t + lane], s);
            atomicAdd(&shacc[128 + 16 * t + lane], q2);
        }
    }
    __syncthreads();
    atomicAdd(&gstats1[tid], shacc[tid]);
}

// BN finalize: scale = gamma * rsqrt(var + eps), shift = beta - mean * scale
__global__ void k_finalize(const float* __restrict__ gstats,
                           const float* __restrict__ gamma,
                           const float* __restrict__ beta,
                           float* __restrict__ scale, float* __restrict__ shift,
                           int nchan, float invcount) {
    const int i = blockIdx.x * blockDim.x + threadIdx.x;
    if (i < nchan) {
        const float mean = gstats[i] * invcount;
        const float var  = gstats[nchan + i] * invcount - mean * mean;
        const float sc   = gamma[i] * rsqrtf(var + EPSV);
        scale[i] = sc;
        shift[i] = beta[i] - mean * sc;
    }
}

// ---------------------------------------------------------------------------
// Kernel 2: recompute GEMM, apply BN1 (bias folded into the shift), gate with
// sigmoid(filter)*softplus(core), reduce over the atom's 12 neighbors
// in-registers (Σ elems + shfl_xor 16), write nbr_sumed[n][c], accumulate
// BN2 stats.  Filter col c pairs with core col c+64 -> tiles t and t+4,
// SAME lane/elem.
// ---------------------------------------------------------------------------
__global__ __launch_bounds__(256) void k_pass2(
        const float* __restrict__ atom, const float* __restrict__ nbrf,
        const int* __restrict__ idx, const _Float16* __restrict__ bswz,
        const float* __restrict__ bias,
        const float* __restrict__ scale1, const float* __restrict__ shift1,
        float* __restrict__ nbr_sumed, float* __restrict__ gstats2) {
    __shared__ _Float16 rowbuf[WPB][16][RSTR];
    __shared__ float shacc[128];  // [0..63]=sum, [64..127]=sumsq
    const int tid = threadIdx.x, lane = tid & 31, w = tid >> 5;
    if (tid < 128) shacc[tid] = 0.f;
    __syncthreads();

    const int gw = blockIdx.x * WPB + w;
    const int nw = gridDim.x * WPB;
    const int hi = lane >> 4;

    // hoisted per-lane BN params with bias folded into the shift:
    // BN(acc + b) = acc*sc + (sh + b*sc)
    float scF[4], shF[4], scC[4], shC[4];
    #pragma unroll
    for (int t = 0; t < 4; ++t) {
        const int cf = 16 * t + (lane & 15);
        scF[t] = scale1[cf];
        shF[t] = shift1[cf] + bias[cf] * scF[t];
        scC[t] = scale1[cf + 64];
        shC[t] = shift1[cf + 64] + bias[cf + 64] * scC[t];
    }
    float rs[4], rq[4];
    #pragma unroll
    for (int t = 0; t < 4; ++t) { rs[t] = 0.f; rq[t] = 0.f; }

    zero_static(rowbuf[w], lane);

    for (int n = gw; n < NATOMS; n += nw) {
        stage_rows(rowbuf[w], atom, nbrf, idx, n, lane);
        asm volatile("s_wait_dscnt 0" ::: "memory");
        v8f acc[8];
        gemm_tile(rowbuf[w], bswz, lane, acc);
        #pragma unroll
        for (int t = 0; t < 4; ++t) {
            float p = 0.f;
            #pragma unroll
            for (int e = 0; e < 8; ++e) {
                const int row = e + 8 * hi;
                if (row < MNBR) {
                    const float vf = acc[t][e]     * scF[t] + shF[t];
                    const float vc = acc[t + 4][e] * scC[t] + shC[t];
                    p += sigmoidf_fast(vf) * softplusf_fast(vc);
                }
            }
            p += __shfl_xor(p, 16);
            if (lane < 16) {
                nbr_sumed[(size_t)n * AF + 16 * t + lane] = p;
                rs[t] += p; rq[t] += p * p;
            }
        }
    }
    #pragma unroll
    for (int t = 0; t < 4; ++t) {
        if (lane < 16) {
            atomicAdd(&shacc[16 * t + lane], rs[t]);
            atomicAdd(&shacc[64 + 16 * t + lane], rq[t]);
        }
    }
    __syncthreads();
    if (tid < 128) atomicAdd(&gstats2[tid], shacc[tid]);
}

// Kernel 3: out = softplus(atom_in_fea + BN2(nbr_sumed))
__global__ void k_out(const float* __restrict__ atom,
                      const float* __restrict__ nbr_sumed,
                      const float* __restrict__ scale2,
                      const float* __restrict__ shift2,
                      float* __restrict__ out) {
    const int i = blockIdx.x * blockDim.x + threadIdx.x;
    if (i < NATOMS * AF) {
        const int c = i & (AF - 1);
        const float v = atom[i] + (nbr_sumed[i] * scale2[c] + shift2[c]);
        out[i] = softplusf(v);
    }
}

// ---------------------------------------------------------------------------
extern "C" void kernel_launch(void* const* d_in, const int* in_sizes, int n_in,
                              void* d_out, int out_size, void* d_ws, size_t ws_size,
                              hipStream_t stream) {
    (void)in_sizes; (void)n_in; (void)out_size; (void)ws_size;
    const float* atom = (const float*)d_in[0];
    const float* nbrf = (const float*)d_in[1];
    const float* W    = (const float*)d_in[2];
    const float* b    = (const float*)d_in[3];
    const float* g1   = (const float*)d_in[4];
    const float* b1   = (const float*)d_in[5];
    const float* g2   = (const float*)d_in[6];
    const float* b2   = (const float*)d_in[7];
    const int*   idx  = (const int*)d_in[8];
    float* out = (float*)d_out;

    char* ws = (char*)d_ws;
    _Float16* bswz   = (_Float16*)(ws);                 // 49152 B
    float* gstats1   = (float*)(ws + 49152);            // 256 f32 (sum|sq)
    float* gstats2   = (float*)(ws + 49152 + 1024);     // 128 f32 (sum|sq)
    float* bn1       = (float*)(ws + 49152 + 1536);     // scale1[128] | shift1[128]
    float* bn2       = (float*)(ws + 49152 + 2560);     // scale2[64]  | shift2[64]
    float* nbrsum    = (float*)(ws + 52224);            // N*64 f32 = 25.6 MB

    k_init<<<96, 256, 0, stream>>>(W, bswz, gstats1, gstats2);
    k_stats1<<<960, 256, 0, stream>>>(atom, nbrf, idx, bswz, b, gstats1);
    k_finalize<<<1, 128, 0, stream>>>(gstats1, g1, b1, bn1, bn1 + 128, 128,
                                      1.f / (float)(NATOMS * MNBR));
    k_pass2<<<960, 256, 0, stream>>>(atom, nbrf, idx, bswz, b,
                                     bn1, bn1 + 128, nbrsum, gstats2);
    k_finalize<<<1, 128, 0, stream>>>(gstats2, g2, b2, bn2, bn2 + 64, 64,
                                      1.f / (float)NATOMS);
    k_out<<<(NATOMS * AF + 255) / 256, 256, 0, stream>>>(atom, nbrsum,
                                                         bn2, bn2 + 64, out);
}
